// ESN_21414706938166
// MI455X (gfx1250) — matrix operand order repaired and checked by reading, not compiled, not run on hardware
//
#include <hip/hip_runtime.h>
#include <hip/hip_bf16.h>
#include <stdint.h>

// ---------------------------------------------------------------------------
// ESN recurrence x_t = tanh(W_in*u_t + W_res @ x_{t-1}) on MI455X (gfx1250).
// Persistent kernel, one wave per 16-row tile, W_res pre-tiled to a WMMA A-operand
// layout in bf16, k = 16h + i per lane half; x broadcast into the B operand so each
// v_wmma_f32_16x16x32_bf16 does 16 rows x 32 k of the GEMV. Device-wide
// sense-reversing barrier between steps.
//
// 64 WGs x 64 threads = 128 waves = 128 row tiles: one WG per WGP halves the
// per-WGP L2 stream (128 KB/step) vs the 32x128 configuration, which is the
// per-step critical path.
// ---------------------------------------------------------------------------

#define ESN_N      2048
#define ESN_NWG    64                 // persistent workgroups
#define ESN_TPB    64                 // 2 waves per WG -> 128 waves total
#define ESN_TILES  (ESN_N / 16)       // 128 row tiles, one per wave
#define ESN_KCH    (ESN_N / 32)       // 64 k-chunks of 32

typedef float v8f   __attribute__((ext_vector_type(8)));
typedef __bf16 v16bf __attribute__((ext_vector_type(16)));

union AUnion { uint4 q[2]; v16bf v; };
union BUnion { uint32_t u[8]; v16bf v; };

// ---------------------------------------------------------------------------
// One-shot: convert W_res (fp32 row-major [N,N]) into bf16 tiles laid out
// per lane in contiguous k order, matched by the B loads:
//   tile m (16 rows), chunk c (32 k), lane l, elem i (16 bf16 per lane)
//   lane l<16  : row = 16m+l,      k_local = {0..15}
//   lane l>=16 : row = 16m+(l-16), k_local = {16..31}
// flat dst index = ((m*64 + c)*32 + l)*16 + i
// ---------------------------------------------------------------------------
__global__ void esn_tile_w(const float* __restrict__ W, __bf16* __restrict__ Wt) {
    unsigned gid = blockIdx.x * blockDim.x + threadIdx.x;   // 0 .. N*N-1
    unsigned i = gid & 15u;
    unsigned l = (gid >> 4) & 31u;
    unsigned c = (gid >> 9) & 63u;
    unsigned m = gid >> 15;
    unsigned row = m * 16u + (l & 15u);
    unsigned hi  = l >> 4;          // 0 or 1
    unsigned grp = i >> 3;          // 0 or 1
    unsigned kl  = i + 16u * hi;
    unsigned k   = c * 32u + kl;
    Wt[gid] = (__bf16)W[(size_t)row * ESN_N + k];
}

// ---------------------------------------------------------------------------
// Persistent recurrence kernel.
// ---------------------------------------------------------------------------
__global__ void __launch_bounds__(ESN_TPB) esn_persistent(
    const float* __restrict__ u,
    const float* __restrict__ Win,
    const __bf16* __restrict__ Wt,
    __bf16* __restrict__ xbuf0,
    __bf16* __restrict__ xbuf1,
    unsigned* __restrict__ barrier_mem,    // [0]=counter, [1]=flag (memset 0)
    float* __restrict__ out,
    int T, int washout)
{
    __shared__ __align__(16) __bf16 xs[ESN_N];              // 4 KB state copy

    const int lane  = threadIdx.x & 31;
    const int waveg = blockIdx.x * (ESN_TPB / 32) + (threadIdx.x >> 5); // tile 0..127

    // This wave's tile stream: 64 chunks * (32 lanes * 16 bf16) = 64 KB.
    const uint4* ap = (const uint4*)(Wt + (size_t)waveg * (ESN_KCH * 512)) + lane * 2;
    const unsigned bbase = (lane & 16) ? 8u : 0u;

    unsigned gen = 0;

    for (int t = 0; t < T; ++t) {
        const __bf16* xprev = (t & 1) ? xbuf1 : xbuf0;
        __bf16*       xnext = (t & 1) ? xbuf0 : xbuf1;

        // ---- stage x_{t-1} (4 KB bf16 = 256 uint4) into LDS ----
        {
            const uint4* src = (const uint4*)xprev;
            uint4*       dst = (uint4*)xs;
            #pragma unroll
            for (int i = threadIdx.x; i < ESN_N / 8; i += ESN_TPB)
                dst[i] = src[i];
        }
        __syncthreads();

        // ---- GEMV tile: 64 x v_wmma_f32_16x16x32_bf16, 2 accumulators ----
        v8f acc0 = {0.f,0.f,0.f,0.f,0.f,0.f,0.f,0.f};
        v8f acc1 = {0.f,0.f,0.f,0.f,0.f,0.f,0.f,0.f};
        const uint32_t* xw = (const uint32_t*)xs;

        #pragma unroll 2
        for (int c = 0; c < ESN_KCH; c += 2) {
            AUnion a0, a1;
            a0.q[0] = ap[(c + 0) * 64 + 0];
            a0.q[1] = ap[(c + 0) * 64 + 1];
            a1.q[0] = ap[(c + 1) * 64 + 0];
            a1.q[1] = ap[(c + 1) * 64 + 1];

            BUnion b0, b1;
            const uint32_t* xp0 = xw + (c + 0) * 16 + bbase;
            const uint32_t* xp1 = xw + (c + 1) * 16 + bbase;
            #pragma unroll
            for (int j = 0; j < 8; ++j) { b0.u[j] = xp0[j]; b1.u[j] = xp1[j]; }

            acc0 = __builtin_amdgcn_wmma_f32_16x16x32_bf16(
                       false, a0.v, false, b0.v, (short)0, acc0, false, false);
            acc1 = __builtin_amdgcn_wmma_f32_16x16x32_bf16(
                       false, a1.v, false, b1.v, (short)0, acc1, false, false);
        }
        v8f acc = acc0 + acc1;

        // ---- epilogue: every column of C equals y; lane0 holds rows 0..7,
        //      lane16 rows 8..15 of this tile in acc[0..7] ----
        if ((lane & 15) == 0) {
            const int rowBase = waveg * 16 + ((lane >> 4) ? 8 : 0);
            const float ut = u[t];
            #pragma unroll
            for (int r = 0; r < 8; ++r) {
                float z = acc[r] + Win[rowBase + r] * ut;
                float h = tanhf(z);
                if (t >= washout)
                    out[(size_t)(t - washout) * ESN_N + rowBase + r] = h;
                xnext[rowBase + r] = (__bf16)h;
            }
        }

        // ---- device-wide sense barrier: publish x_t, wait for all WGs ----
        ++gen;
        __threadfence();            // each thread's stores visible at agent scope
        __syncthreads();
        if (threadIdx.x == 0) {
            unsigned old = __hip_atomic_fetch_add(&barrier_mem[0], 1u,
                               __ATOMIC_ACQ_REL, __HIP_MEMORY_SCOPE_AGENT);
            if (old == (unsigned)ESN_NWG - 1) {
                __hip_atomic_store(&barrier_mem[0], 0u,
                                   __ATOMIC_RELAXED, __HIP_MEMORY_SCOPE_AGENT);
                __hip_atomic_store(&barrier_mem[1], gen,
                                   __ATOMIC_RELEASE, __HIP_MEMORY_SCOPE_AGENT);
            } else {
                while (__hip_atomic_load(&barrier_mem[1],
                           __ATOMIC_ACQUIRE, __HIP_MEMORY_SCOPE_AGENT) < gen) {
                    __builtin_amdgcn_s_sleep(1);
                }
            }
        }
        __syncthreads();
    }
}

// ---------------------------------------------------------------------------
// Host launch. Inputs: [0]=u (T f32), [1]=W_in (N f32), [2]=W_res (N*N f32),
// [3]=washout (int, unused on host: recovered from out_size). Output fp32
// (T-washout)*N. Scratch: barrier(256B) | xbuf0(4KB) | xbuf1(4KB) | Wt(8MB).
// ---------------------------------------------------------------------------
extern "C" void kernel_launch(void* const* d_in, const int* in_sizes, int n_in,
                              void* d_out, int out_size, void* d_ws, size_t ws_size,
                              hipStream_t stream) {
    const float* u    = (const float*)d_in[0];
    const float* Win  = (const float*)d_in[1];
    const float* Wres = (const float*)d_in[2];
    (void)n_in;

    const int T = in_sizes[0];
    const int washout = T - out_size / ESN_N;

    unsigned char* ws = (unsigned char*)d_ws;
    unsigned* bar   = (unsigned*)ws;                       // 256 B region
    __bf16*   xbuf0 = (__bf16*)(ws + 256);                 // 4 KB
    __bf16*   xbuf1 = (__bf16*)(ws + 256 + 4096);          // 4 KB
    __bf16*   Wt    = (__bf16*)(ws + 16384);               // 8 MB tiled W
    (void)ws_size;

    // Zero barrier state + x_{-1} = 0 (bf16 zero is 0x0000). Graph-capturable.
    hipMemsetAsync(ws, 0, 256 + 4096, stream);

    // One-shot W tiling: N*N elements, 256 threads/block.
    esn_tile_w<<<dim3((ESN_N * ESN_N) / 256), dim3(256), 0, stream>>>(Wres, Wt);

    // Persistent recurrence: 64 WGs x 64 threads = 128 waves = 128 row tiles.
    esn_persistent<<<dim3(ESN_NWG), dim3(ESN_TPB), 0, stream>>>(
        u, Win, Wt, xbuf0, xbuf1, bar, (float*)d_out, T, washout);
}
